// NSM_29222957482522
// MI455X (gfx1250) — matrix operand order repaired and checked by reading, not compiled, not run on hardware
//
#include <hip/hip_runtime.h>

// ---------------------------------------------------------------------------
// Problem constants (from reference)
// ---------------------------------------------------------------------------
#define BB    64
#define NPG   160
#define PP    16
#define HH    300
#define DEG   16
#define II    5
#define OUTD  2000
#define NN    (BB*NPG)        // 10240 nodes
#define EE    (NN*DEG)        // 163840 edges
#define DD    (PP+1)          // 17
#define KNODE (PP*HH)         // 4800
#define NOUTP 320             // padded output cols (multiple of 32, 16B align)
#define EPB   (NPG*DEG)       // 2560 edges per batch

typedef __attribute__((ext_vector_type(16))) __bf16 bf16x16;
typedef __attribute__((ext_vector_type(8)))  float  floatx8;

union Frag16 { bf16x16 v; unsigned short u[16]; unsigned int d[8]; };

static __device__ __forceinline__ unsigned short f32_to_bf16(float f) {
  unsigned u = __float_as_uint(f);
  unsigned r = u + 0x7FFFu + ((u >> 16) & 1u);   // round-to-nearest-even
  return (unsigned short)(r >> 16);
}

static __device__ __forceinline__ unsigned int pack_bf16(float lo, float hi) {
  return (unsigned)f32_to_bf16(lo) | ((unsigned)f32_to_bf16(hi) << 16);
}

static __device__ __forceinline__ float elu1(float x) {
  return x > 0.0f ? x : (__expf(x) - 1.0f);
}

static __device__ __forceinline__ void load_bfrag(const unsigned short* p, Frag16& f) {
#pragma unroll
  for (int s = 0; s < 16; ++s) f.u[s] = p[(size_t)(s < 8 ? s : s + 8) * NOUTP];
}

// Reduce one 16x16 accumulator tile: out[row0+m] += sum_n elu(C[m,n])*wvec[tile_n0+n]
static __device__ __forceinline__ void reduce_store(floatx8 acc, int tile_n0, int col, int half,
                                                    const float* __restrict__ wvec,
                                                    float* __restrict__ outv, int row0) {
  int nn = tile_n0 + col;
  float w = (nn < HH) ? wvec[nn] : 0.f;
#pragma unroll
  for (int r = 0; r < 8; ++r) {
    float part = elu1(acc[r]) * w;
    part += __shfl_xor(part, 1);
    part += __shfl_xor(part, 2);
    part += __shfl_xor(part, 4);
    part += __shfl_xor(part, 8);
    if (col == 0) atomicAdd(&outv[row0 + r + 8 * half], part);
  }
}

// ---------------------------------------------------------------------------
// sims[b,i,d] = softmax_d( instructions[b,i,:] . property_embeddings[d,:] )
// ---------------------------------------------------------------------------
__global__ void k_sims(const float* __restrict__ instr, const float* __restrict__ pe,
                       float* __restrict__ sims) {
  int bi = blockIdx.x;                 // b*II + i
  int d = threadIdx.x;
  __shared__ float sh[DD];
  if (d < DD) {
    const float* ir = instr + (size_t)bi * HH;
    const float* pr = pe + (size_t)d * HH;
    float acc = 0.f;
    for (int h = 0; h < HH; ++h) acc += ir[h] * pr[h];
    sh[d] = acc;
  }
  __syncthreads();
  if (d == 0) {
    float m = sh[0];
    for (int j = 1; j < DD; ++j) m = fmaxf(m, sh[j]);
    float s = 0.f;
    for (int j = 0; j < DD; ++j) { sh[j] = __expf(sh[j] - m); s += sh[j]; }
    float inv = 1.f / s;
    for (int j = 0; j < DD; ++j) sims[(size_t)bi * DD + j] = sh[j] * inv;
  }
}

// ---------------------------------------------------------------------------
// f32 [srcRows x srcCols] -> bf16 [* x dstCols], zero-padded
// ---------------------------------------------------------------------------
__global__ void k_conv(const float* __restrict__ src, unsigned short* __restrict__ dst,
                       int srcRows, int srcCols, int dstCols, int total) {
  int i = blockIdx.x * 256 + threadIdx.x;
  if (i >= total) return;
  int r = i / dstCols, c = i - r * dstCols;
  float v = (r < srcRows && c < srcCols) ? src[(size_t)r * srcCols + c] : 0.f;
  dst[i] = f32_to_bf16(v);
}

__global__ void k_zero(float* __restrict__ p, int n) {
  int i = blockIdx.x * 256 + threadIdx.x;
  if (i < n) p[i] = 0.f;
}

__global__ void k_init_dist(const float* __restrict__ npg, float* __restrict__ dist) {
  int n = blockIdx.x * 256 + threadIdx.x;
  if (n < NN) dist[n] = 1.0f / npg[n / NPG];
}

// ---------------------------------------------------------------------------
// gate[b][k] = sims[b,step,p] * instr[b,step,h],  k = p*HH + h
// ---------------------------------------------------------------------------
__global__ void k_gate(const float* __restrict__ sims, const float* __restrict__ instr,
                       float* __restrict__ gate, int step) {
  int i = blockIdx.x * 256 + threadIdx.x;
  if (i >= BB * KNODE) return;
  int b = i / KNODE, k = i - b * KNODE;
  int p = k / HH, h = k - p * HH;
  gate[i] = sims[((size_t)b * II + step) * DD + p] * instr[((size_t)b * II + step) * HH + h];
}

// ---------------------------------------------------------------------------
// Node GEMM: sstate[n] += sum_k elu( (gated A @ Wp)[n,k] ) * w_node[k]
// 640 blocks (16 rows), 5 waves/block, each wave owns 4 output tiles.
// A fragments loaded directly from global (two 32B runs/lane), gated+cvt'd
// in registers -> 4 back-to-back WMMAs per fragment. No LDS, no barriers.
// ---------------------------------------------------------------------------
__global__ __launch_bounds__(160)
void k_node_gemm(const float* __restrict__ attrs, const float* __restrict__ gate,
                 const unsigned short* __restrict__ Wp, const float* __restrict__ wnode,
                 float* __restrict__ sstate) {
  const int tid  = threadIdx.x;
  const int n0   = blockIdx.x * 16;
  const int b    = blockIdx.x / (NPG / 16);
  const int lane = tid & 31;
  const int wv   = tid >> 5;          // 0..4
  const int half = lane >> 4;
  const int col  = lane & 15;         // A row / C column
  const int c0   = half * 8;

  const float* gb   = gate + (size_t)b * KNODE;
  const float* arow = attrs + (size_t)(n0 + col) * KNODE;

  const int t0 = (wv + 0) * 16, t1 = (wv + 5) * 16, t2 = (wv + 10) * 16, t3 = (wv + 15) * 16;
  floatx8 acc0 = {}, acc1 = {}, acc2 = {}, acc3 = {};

  for (int kt = 0; kt < KNODE / 32; ++kt) {
    const int k0 = kt * 32;
    Frag16 a;
    {
      const float4* ap0 = (const float4*)(arow + k0 + c0);
      const float4* ap1 = (const float4*)(arow + k0 + 16 + c0);
      const float4* gp0 = (const float4*)(gb + k0 + c0);
      const float4* gp1 = (const float4*)(gb + k0 + 16 + c0);
      float4 A0 = ap0[0], A1 = ap0[1], A2 = ap1[0], A3 = ap1[1];
      float4 G0 = gp0[0], G1 = gp0[1], G2 = gp1[0], G3 = gp1[1];
      a.d[0] = pack_bf16(A0.x * G0.x, A0.y * G0.y);
      a.d[1] = pack_bf16(A0.z * G0.z, A0.w * G0.w);
      a.d[2] = pack_bf16(A1.x * G1.x, A1.y * G1.y);
      a.d[3] = pack_bf16(A1.z * G1.z, A1.w * G1.w);
      a.d[4] = pack_bf16(A2.x * G2.x, A2.y * G2.y);
      a.d[5] = pack_bf16(A2.z * G2.z, A2.w * G2.w);
      a.d[6] = pack_bf16(A3.x * G3.x, A3.y * G3.y);
      a.d[7] = pack_bf16(A3.z * G3.z, A3.w * G3.w);
    }
    const unsigned short* wr = Wp + (size_t)(k0 + c0) * NOUTP + col;
    Frag16 b0, b1, b2, b3;
    load_bfrag(wr + t0, b0);
    load_bfrag(wr + t1, b1);
    load_bfrag(wr + t2, b2);
    load_bfrag(wr + t3, b3);
    acc0 = __builtin_amdgcn_wmma_f32_16x16x32_bf16(false, a.v, false, b0.v, (short)0, acc0, false, false);
    acc1 = __builtin_amdgcn_wmma_f32_16x16x32_bf16(false, a.v, false, b1.v, (short)0, acc1, false, false);
    acc2 = __builtin_amdgcn_wmma_f32_16x16x32_bf16(false, a.v, false, b2.v, (short)0, acc2, false, false);
    acc3 = __builtin_amdgcn_wmma_f32_16x16x32_bf16(false, a.v, false, b3.v, (short)0, acc3, false, false);
  }

  reduce_store(acc0, t0, col, half, wnode, sstate, n0);
  reduce_store(acc1, t1, col, half, wnode, sstate, n0);
  reduce_store(acc2, t2, col, half, wnode, sstate, n0);
  reduce_store(acc3, t3, col, half, wnode, sstate, n0);
}

// ---------------------------------------------------------------------------
// Edge GEMM: t[e] += sum_h elu( ((instr[b]*edge_attrs[e]) @ We)[e,h] ) * w_rel[h]
// Same structure; 9 unguarded k-tiles + guarded tail (K 288..319).
// ---------------------------------------------------------------------------
__global__ __launch_bounds__(160)
void k_edge_gemm(const float* __restrict__ eattr, const float* __restrict__ instr,
                 const unsigned short* __restrict__ We, const float* __restrict__ wrel,
                 float* __restrict__ tvec, int step) {
  const int tid  = threadIdx.x;
  const int e0   = blockIdx.x * 16;
  const int b    = blockIdx.x / (EPB / 16);
  const int lane = tid & 31;
  const int wv   = tid >> 5;
  const int half = lane >> 4;
  const int col  = lane & 15;
  const int c0   = half * 8;

  const float* insrow = instr + ((size_t)b * II + step) * HH;
  const float* erow   = eattr + (size_t)(e0 + col) * HH;

  const int t0 = (wv + 0) * 16, t1 = (wv + 5) * 16, t2 = (wv + 10) * 16, t3 = (wv + 15) * 16;
  floatx8 acc0 = {}, acc1 = {}, acc2 = {}, acc3 = {};

  for (int kt = 0; kt < 9; ++kt) {           // K 0..287, fully in-bounds
    const int k0 = kt * 32;
    Frag16 a;
    {
      const float4* ap0 = (const float4*)(erow + k0 + c0);
      const float4* ap1 = (const float4*)(erow + k0 + 16 + c0);
      const float4* gp0 = (const float4*)(insrow + k0 + c0);
      const float4* gp1 = (const float4*)(insrow + k0 + 16 + c0);
      float4 A0 = ap0[0], A1 = ap0[1], A2 = ap1[0], A3 = ap1[1];
      float4 G0 = gp0[0], G1 = gp0[1], G2 = gp1[0], G3 = gp1[1];
      a.d[0] = pack_bf16(A0.x * G0.x, A0.y * G0.y);
      a.d[1] = pack_bf16(A0.z * G0.z, A0.w * G0.w);
      a.d[2] = pack_bf16(A1.x * G1.x, A1.y * G1.y);
      a.d[3] = pack_bf16(A1.z * G1.z, A1.w * G1.w);
      a.d[4] = pack_bf16(A2.x * G2.x, A2.y * G2.y);
      a.d[5] = pack_bf16(A2.z * G2.z, A2.w * G2.w);
      a.d[6] = pack_bf16(A3.x * G3.x, A3.y * G3.y);
      a.d[7] = pack_bf16(A3.z * G3.z, A3.w * G3.w);
    }
    const unsigned short* wr = We + (size_t)(k0 + c0) * NOUTP + col;
    Frag16 b0, b1, b2, b3;
    load_bfrag(wr + t0, b0);
    load_bfrag(wr + t1, b1);
    load_bfrag(wr + t2, b2);
    load_bfrag(wr + t3, b3);
    acc0 = __builtin_amdgcn_wmma_f32_16x16x32_bf16(false, a.v, false, b0.v, (short)0, acc0, false, false);
    acc1 = __builtin_amdgcn_wmma_f32_16x16x32_bf16(false, a.v, false, b1.v, (short)0, acc1, false, false);
    acc2 = __builtin_amdgcn_wmma_f32_16x16x32_bf16(false, a.v, false, b2.v, (short)0, acc2, false, false);
    acc3 = __builtin_amdgcn_wmma_f32_16x16x32_bf16(false, a.v, false, b3.v, (short)0, acc3, false, false);
  }
  {                                          // tail: K 288..319, guard k<300
    const int k0 = 288;
    Frag16 a;
#pragma unroll
    for (int j = 0; j < 8; ++j) {
      int k = k0 + c0 + j;
      float v = (k < HH) ? erow[k] * insrow[k] : 0.f;
      a.u[j] = f32_to_bf16(v);
      a.u[8 + j] = 0;                        // K 304..319 -> zero
    }
    const unsigned short* wr = We + (size_t)(k0 + c0) * NOUTP + col;
    Frag16 b0, b1, b2, b3;
    load_bfrag(wr + t0, b0);
    load_bfrag(wr + t1, b1);
    load_bfrag(wr + t2, b2);
    load_bfrag(wr + t3, b3);
    acc0 = __builtin_amdgcn_wmma_f32_16x16x32_bf16(false, a.v, false, b0.v, (short)0, acc0, false, false);
    acc1 = __builtin_amdgcn_wmma_f32_16x16x32_bf16(false, a.v, false, b1.v, (short)0, acc1, false, false);
    acc2 = __builtin_amdgcn_wmma_f32_16x16x32_bf16(false, a.v, false, b2.v, (short)0, acc2, false, false);
    acc3 = __builtin_amdgcn_wmma_f32_16x16x32_bf16(false, a.v, false, b3.v, (short)0, acc3, false, false);
  }

  reduce_store(acc0, t0, col, half, wrel, tvec, e0);
  reduce_store(acc1, t1, col, half, wrel, tvec, e0);
  reduce_store(acc2, t2, col, half, wrel, tvec, e0);
  reduce_store(acc3, t3, col, half, wrel, tvec, e0);
}

// ---------------------------------------------------------------------------
// srel[dst] += dist[src] * t[e]   (msg . w_rel distributes over edges)
// ---------------------------------------------------------------------------
__global__ void k_scatter(const float* __restrict__ t, const float* __restrict__ dist,
                          const int* __restrict__ esrc, const int* __restrict__ edst,
                          float* __restrict__ srel) {
  int e = blockIdx.x * 256 + threadIdx.x;
  if (e >= EE) return;
  atomicAdd(&srel[edst[e]], dist[esrc[e]] * t[e]);
}

// ---------------------------------------------------------------------------
// Per-graph segment softmaxes + dist update
// ---------------------------------------------------------------------------
__global__ void k_update(const float* __restrict__ sstate, const float* __restrict__ srel,
                         const float* __restrict__ sims, float* __restrict__ dist, int step) {
  int b = blockIdx.x;
  int i = threadIdx.x;                 // 0..159
  int n = b * NPG + i;
  __shared__ float sh[NPG];
  __shared__ float red[2];

  float ss = sstate[n];
  sh[i] = ss; __syncthreads();
  if (i == 0) { float m = -1e30f; for (int j = 0; j < NPG; ++j) m = fmaxf(m, sh[j]); red[0] = m; }
  __syncthreads();
  float es = __expf(ss - red[0]);
  sh[i] = es; __syncthreads();
  if (i == 0) { float s = 0.f; for (int j = 0; j < NPG; ++j) s += sh[j]; red[1] = s; }
  __syncthreads();
  float nd_state = es / red[1];
  __syncthreads();

  float sr = srel[n];
  sh[i] = sr; __syncthreads();
  if (i == 0) { float m = -1e30f; for (int j = 0; j < NPG; ++j) m = fmaxf(m, sh[j]); red[0] = m; }
  __syncthreads();
  float er = __expf(sr - red[0]);
  sh[i] = er; __syncthreads();
  if (i == 0) { float s = 0.f; for (int j = 0; j < NPG; ++j) s += sh[j]; red[1] = s; }
  __syncthreads();
  float nd_rel = er / red[1];

  float r = sims[((size_t)b * II + step) * DD + (DD - 1)];
  dist[n] = r * nd_rel + (1.f - r) * nd_state;
}

// ---------------------------------------------------------------------------
// agg[b,h] = sum_n dist[n] * sum_p sims[b,last,p] * node_attrs[n,p,h]
// ---------------------------------------------------------------------------
__global__ void k_aggregate(const float* __restrict__ attrs, const float* __restrict__ sims,
                            const float* __restrict__ dist, float* __restrict__ agg) {
  int b = blockIdx.x;
  int h = threadIdx.x;                 // 0..319
  __shared__ float pl[PP];
  if (threadIdx.x < PP) pl[threadIdx.x] = sims[((size_t)b * II + (II - 1)) * DD + threadIdx.x];
  __syncthreads();
  if (h >= HH) return;
  float acc = 0.f;
  for (int j = 0; j < NPG; ++j) {
    int n = b * NPG + j;
    float d = dist[n];
    const float* base = attrs + ((size_t)n * PP) * HH + h;
    float t = 0.f;
#pragma unroll
    for (int p = 0; p < PP; ++p) t += pl[p] * base[(size_t)p * HH];
    acc += d * t;
  }
  agg[(size_t)b * HH + h] = acc;
}

// ---------------------------------------------------------------------------
// z = elu([encq, agg] @ W1 + b1);   out = z @ W2 + b2
// ---------------------------------------------------------------------------
__global__ void k_mlp1(const float* __restrict__ encq, const float* __restrict__ agg,
                       const float* __restrict__ W1, const float* __restrict__ b1,
                       float* __restrict__ z) {
  int b = blockIdx.x;
  int j = threadIdx.x;
  if (j >= 2 * HH) return;
  float acc = b1[j];
  const float* eq = encq + (size_t)b * HH;
  const float* ag = agg + (size_t)b * HH;
  for (int k = 0; k < HH; ++k) acc += eq[k] * W1[(size_t)k * (2 * HH) + j];
  for (int k = 0; k < HH; ++k) acc += ag[k] * W1[(size_t)(HH + k) * (2 * HH) + j];
  z[(size_t)b * (2 * HH) + j] = elu1(acc);
}

__global__ void k_mlp2(const float* __restrict__ z, const float* __restrict__ W2,
                       const float* __restrict__ b2, float* __restrict__ out) {
  int b = blockIdx.y;
  int o = blockIdx.x * 256 + threadIdx.x;
  if (o >= OUTD) return;
  float acc = b2[o];
  const float* zr = z + (size_t)b * (2 * HH);
  for (int k = 0; k < 2 * HH; ++k) acc += zr[k] * W2[(size_t)k * OUTD + o];
  out[(size_t)b * OUTD + o] = acc;
}

// ---------------------------------------------------------------------------
extern "C" void kernel_launch(void* const* d_in, const int* in_sizes, int n_in,
                              void* d_out, int out_size, void* d_ws, size_t ws_size,
                              hipStream_t stream) {
  const float* node_attrs   = (const float*)d_in[0];
  const float* edge_attrs   = (const float*)d_in[1];
  const float* instructions = (const float*)d_in[2];
  const float* encq         = (const float*)d_in[3];
  const float* prop_emb     = (const float*)d_in[4];
  const float* npg          = (const float*)d_in[5];
  const int*   esrc         = (const int*)d_in[8];
  const int*   edst         = (const int*)d_in[9];
  const float* W_props      = (const float*)d_in[10];
  const float* W_edge       = (const float*)d_in[11];
  const float* w_node       = (const float*)d_in[12];
  const float* w_rel        = (const float*)d_in[13];
  const float* W1           = (const float*)d_in[14];
  const float* b1           = (const float*)d_in[15];
  const float* W2           = (const float*)d_in[16];
  const float* b2           = (const float*)d_in[17];
  float* out = (float*)d_out;

  char* ws = (char*)d_ws;
  float*          sims   = (float*)ws;          ws += 21760;    // B*I*D f32
  unsigned short* Wp     = (unsigned short*)ws; ws += 3072000;  // 4800x320 bf16
  unsigned short* We     = (unsigned short*)ws; ws += 204800;   // 320x320 bf16
  float*          gate   = (float*)ws;          ws += 1228800;  // B x 4800 f32
  float*          sstate = (float*)ws;          ws += 40960;    // N f32
  float*          srel   = (float*)ws;          ws += 40960;    // N f32
  float*          tvec   = (float*)ws;          ws += 655360;   // E f32
  float*          dist   = (float*)ws;          ws += 40960;    // N f32
  float*          agg    = (float*)ws;          ws += 76800;    // B*H f32
  float*          z      = (float*)ws;          ws += 153600;   // B*2H f32

  k_sims<<<BB * II, 32, 0, stream>>>(instructions, prop_emb, sims);
  k_conv<<<(KNODE * NOUTP + 255) / 256, 256, 0, stream>>>(W_props, Wp, KNODE, HH, NOUTP, KNODE * NOUTP);
  k_conv<<<(NOUTP * NOUTP + 255) / 256, 256, 0, stream>>>(W_edge, We, HH, HH, NOUTP, NOUTP * NOUTP);
  k_init_dist<<<(NN + 255) / 256, 256, 0, stream>>>(npg, dist);

  for (int step = 0; step < II - 1; ++step) {
    k_zero<<<(2 * NN + EE + 255) / 256, 256, 0, stream>>>(sstate, 2 * NN + EE); // sstate|srel|tvec contiguous
    k_gate<<<(BB * KNODE + 255) / 256, 256, 0, stream>>>(sims, instructions, gate, step);
    k_node_gemm<<<NN / 16, 160, 0, stream>>>(node_attrs, gate, Wp, w_node, sstate);
    k_edge_gemm<<<EE / 16, 160, 0, stream>>>(edge_attrs, instructions, We, w_rel, tvec, step);
    k_scatter<<<(EE + 255) / 256, 256, 0, stream>>>(tvec, dist, esrc, edst, srel);
    k_update<<<BB, NPG, 0, stream>>>(sstate, srel, sims, dist, step);
  }

  k_aggregate<<<BB, NOUTP, 0, stream>>>(node_attrs, sims, dist, agg);
  k_mlp1<<<BB, 640, 0, stream>>>(encq, agg, W1, b1, z);
  k_mlp2<<<dim3((OUTD + 255) / 256, BB), 256, 0, stream>>>(z, W2, b2, out);
}
